// GGSNN_16260746183346
// MI455X (gfx1250) — matrix operand order repaired and checked by reading, not compiled, not run on hardware
//
#include <hip/hip_runtime.h>
#include <hip/hip_bf16.h>
#include <math.h>

typedef __attribute__((ext_vector_type(16))) _Float16 v16h;
typedef __attribute__((ext_vector_type(8)))  float    v8f;

#define NODES 64
#define DIM   128

// ---------- prep: fp32 weights -> f16 in workspace (L2-resident) ----------
__global__ void convert_f32_to_f16(const float* __restrict__ src,
                                   _Float16* __restrict__ dst, int n) {
  int i = blockIdx.x * blockDim.x + threadIdx.x;
  if (i < n) dst[i] = (_Float16)src[i];
}

// ---------- WMMA fragment loaders (16x16x32 f16; contiguous k order, same for A and B) ----------
// A 16x32 (MxK): lane m = lane&15; lanes 0-15 hold K=0..15,
// lanes 16-31 hold K=16..31. Row-major source, k contiguous.
__device__ __forceinline__ v16h load_fragA(const _Float16* src, int lda,
                                           int m0, int k0, int lane) {
  const int m    = m0 + (lane & 15);
  const int half = lane >> 4;
  const _Float16* p = src + m * lda + k0 + half * 16;
  v16h a;
#pragma unroll
  for (int e = 0; e < 16; ++e) a[e] = p[e];       // 2x b128
  return a;
}

// B 32x16 (KxN): B[k][n] = src_t[n*ldt + k] (k contiguous per lane).
// lane n = lane&15; lanes 0-15 K=0..15, lanes 16-31 K=16..31
__device__ __forceinline__ v16h load_fragB(const _Float16* src_t, int ldt,
                                           int k0, int n0, int lane) {
  const int n  = n0 + (lane & 15);
  const int kb = k0 + (lane >> 4) * 16;
  const _Float16* p = src_t + n * ldt + kb;
  v16h b;
#pragma unroll
  for (int e = 0; e < 16; ++e) b[e] = p[e];       // 2x b128
  return b;
}

__device__ __forceinline__ void zero_acc(v8f acc[4]) {
#pragma unroll
  for (int nt = 0; nt < 4; ++nt)
#pragma unroll
    for (int v = 0; v < 8; ++v) acc[nt][v] = 0.0f;
}

__device__ __forceinline__ float sigm(float x) {
  return 1.0f / (1.0f + __expf(-x));
}

// ---------- fused GGS-NN: one workgroup per graph, 4 steps in LDS ----------
// All GEMMs computed transposed: Y^T = W @ X^T.
//   M = output-dim (wave w owns m-tile m0=16w), N = node (4 n-tiles), K = in-dim.
//   A-frag  = weight rows (global, k-contiguous)
//   B-frag  = row-major activation in LDS (k-contiguous)
//   C-tile  = row-major activation [node][dim], contiguous in v -> b128 stores
__global__ __launch_bounds__(256)
__attribute__((amdgpu_num_vgpr(256))) void ggsnn_fused(
    const float* __restrict__ h_in, const float* __restrict__ mask,
    const _Float16* __restrict__ wf,
    const float* __restrict__ bp, const float* __restrict__ bz,
    const float* __restrict__ buz, const float* __restrict__ br,
    const float* __restrict__ bur, const float* __restrict__ bh,
    const float* __restrict__ buh, float* __restrict__ h_out) {
  __shared__ _Float16 sh_h[NODES * DIM];      // 16 KB  h row-major [node][dim]
  __shared__ _Float16 sh_hT[DIM * NODES];     // 16 KB  h transposed [dim][node]
  __shared__ _Float16 sh_neigh[NODES * DIM];  // 16 KB  neigh row-major
  __shared__ _Float16 sh_tmp[NODES * DIM];    // 16 KB  agg / r*h row-major
  __shared__ _Float16 sh_z[NODES * DIM];      // 16 KB  z gate row-major
  __shared__ _Float16 sh_mask[NODES * NODES]; //  8 KB  mask row-major [i][j]
  __shared__ float    sh_bias[4 * DIM];       //  2 KB  combined biases

  const int tid   = threadIdx.x;
  const int lane  = tid & 31;
  const int wv    = tid >> 5;       // 8 waves
  const int m0    = wv * 16;        // this wave's output-dim tile
  const int batch = blockIdx.x;

  const float* hb = h_in + (size_t)batch * NODES * DIM;
  const float* mb = mask + (size_t)batch * NODES * NODES;

#pragma unroll 4
  for (int i = tid; i < NODES * DIM; i += 256) {
    _Float16 v = (_Float16)hb[i];
    int node = i >> 7, d = i & (DIM - 1);
    sh_h[i] = v;
    sh_hT[d * NODES + node] = v;
  }
#pragma unroll 4
  for (int i = tid; i < NODES * NODES; i += 256) sh_mask[i] = (_Float16)mb[i];
  // combined biases -> LDS (short-lived VGPR use at consumption points)
  for (int i = tid; i < DIM; i += 256) {
    sh_bias[0 * DIM + i] = 64.0f * bp[i];  // reference scales W_p bias by n
    sh_bias[1 * DIM + i] = bz[i] + buz[i];
    sh_bias[2 * DIM + i] = br[i] + bur[i];
    sh_bias[3 * DIM + i] = bh[i] + buh[i];
  }

  const _Float16* Wp = wf + 0 * DIM * DIM;
  const _Float16* Wz = wf + 1 * DIM * DIM;
  const _Float16* Uz = wf + 2 * DIM * DIM;
  const _Float16* Wr = wf + 3 * DIM * DIM;
  const _Float16* Ur = wf + 4 * DIM * DIM;
  const _Float16* Wh = wf + 5 * DIM * DIM;
  const _Float16* Uh = wf + 6 * DIM * DIM;

  const int nlo   = lane & 15;        // node-column within n-tile
  const int mbase = (lane >> 4) * 8;  // dim-row offset within m-tile
  const int mrow  = m0 + mbase;       // first dim-row this lane holds
  __syncthreads();

  v8f acc[4], acc2[4];

#pragma unroll 1
  for (int step = 0; step < 4; ++step) {
    // 1) agg^T = h^T @ mask^T  -> sh_tmp (row-major agg[node][dim])
    zero_acc(acc);
#pragma unroll
    for (int k0 = 0; k0 < NODES; k0 += 32) {
      v16h a = load_fragA(sh_hT, NODES, m0, k0, lane);
      v16h b[4];
#pragma unroll
      for (int nt = 0; nt < 4; ++nt)
        b[nt] = load_fragB(sh_mask, NODES, k0, nt * 16, lane);
#pragma unroll
      for (int nt = 0; nt < 4; ++nt)
        acc[nt] = __builtin_amdgcn_wmma_f32_16x16x32_f16(
            false, a, false, b[nt], (short)0, acc[nt], false, false);
    }
#pragma unroll
    for (int nt = 0; nt < 4; ++nt) {
      _Float16* p = sh_tmp + (nt * 16 + nlo) * DIM + mrow;
#pragma unroll
      for (int v = 0; v < 8; ++v) p[v] = (_Float16)acc[nt][v];  // b128
    }
    __syncthreads();

    // 2) neigh^T = Wp @ agg^T + 64*bp -> sh_neigh (row-major)
    zero_acc(acc);
#pragma unroll
    for (int k0 = 0; k0 < DIM; k0 += 32) {
      v16h a = load_fragA(Wp, DIM, m0, k0, lane);
      v16h b[4];
#pragma unroll
      for (int nt = 0; nt < 4; ++nt)
        b[nt] = load_fragB(sh_tmp, DIM, k0, nt * 16, lane);
#pragma unroll
      for (int nt = 0; nt < 4; ++nt)
        acc[nt] = __builtin_amdgcn_wmma_f32_16x16x32_f16(
            false, a, false, b[nt], (short)0, acc[nt], false, false);
    }
    {
      float bv[8];
#pragma unroll
      for (int v = 0; v < 8; ++v) bv[v] = sh_bias[0 * DIM + mrow + v];
#pragma unroll
      for (int nt = 0; nt < 4; ++nt) {
        _Float16* p = sh_neigh + (nt * 16 + nlo) * DIM + mrow;
#pragma unroll
        for (int v = 0; v < 8; ++v) p[v] = (_Float16)(acc[nt][v] + bv[v]);
      }
    }
    __syncthreads();

    // 3) fused z/r: z^T = Wz@neigh^T + Uz@h^T ; r^T = Wr@neigh^T + Ur@h^T
    //    B-fragment shared by both products.
    zero_acc(acc);   // z
    zero_acc(acc2);  // r
#pragma unroll
    for (int k0 = 0; k0 < DIM; k0 += 32) {
      v16h az = load_fragA(Wz, DIM, m0, k0, lane);
      v16h ar = load_fragA(Wr, DIM, m0, k0, lane);
#pragma unroll
      for (int nt = 0; nt < 4; ++nt) {
        v16h b = load_fragB(sh_neigh, DIM, k0, nt * 16, lane);
        acc[nt] = __builtin_amdgcn_wmma_f32_16x16x32_f16(
            false, az, false, b, (short)0, acc[nt], false, false);
        acc2[nt] = __builtin_amdgcn_wmma_f32_16x16x32_f16(
            false, ar, false, b, (short)0, acc2[nt], false, false);
      }
    }
#pragma unroll
    for (int k0 = 0; k0 < DIM; k0 += 32) {
      v16h az = load_fragA(Uz, DIM, m0, k0, lane);
      v16h ar = load_fragA(Ur, DIM, m0, k0, lane);
#pragma unroll
      for (int nt = 0; nt < 4; ++nt) {
        v16h b = load_fragB(sh_h, DIM, k0, nt * 16, lane);
        acc[nt] = __builtin_amdgcn_wmma_f32_16x16x32_f16(
            false, az, false, b, (short)0, acc[nt], false, false);
        acc2[nt] = __builtin_amdgcn_wmma_f32_16x16x32_f16(
            false, ar, false, b, (short)0, acc2[nt], false, false);
      }
    }
    {
      float bvz[8], bvr[8];
#pragma unroll
      for (int v = 0; v < 8; ++v) {
        bvz[v] = sh_bias[1 * DIM + mrow + v];
        bvr[v] = sh_bias[2 * DIM + mrow + v];
      }
#pragma unroll
      for (int nt = 0; nt < 4; ++nt) {
        int base = (nt * 16 + nlo) * DIM + mrow;
#pragma unroll
        for (int v = 0; v < 8; ++v) {
          sh_z[base + v] = (_Float16)sigm(acc[nt][v] + bvz[v]);
          float r = sigm(acc2[nt][v] + bvr[v]);
          sh_tmp[base + v] = (_Float16)(r * (float)sh_h[base + v]);
        }
      }
    }
    __syncthreads();

    // 4) hhat^T = Wh@neigh^T + Uh@(r*h)^T ; h = (1-z)h + z*tanh(hhat)
    zero_acc(acc);
#pragma unroll
    for (int k0 = 0; k0 < DIM; k0 += 32) {
      v16h a = load_fragA(Wh, DIM, m0, k0, lane);
      v16h b[4];
#pragma unroll
      for (int nt = 0; nt < 4; ++nt)
        b[nt] = load_fragB(sh_neigh, DIM, k0, nt * 16, lane);
#pragma unroll
      for (int nt = 0; nt < 4; ++nt)
        acc[nt] = __builtin_amdgcn_wmma_f32_16x16x32_f16(
            false, a, false, b[nt], (short)0, acc[nt], false, false);
    }
#pragma unroll
    for (int k0 = 0; k0 < DIM; k0 += 32) {
      v16h a = load_fragA(Uh, DIM, m0, k0, lane);
      v16h b[4];
#pragma unroll
      for (int nt = 0; nt < 4; ++nt)
        b[nt] = load_fragB(sh_tmp, DIM, k0, nt * 16, lane);
#pragma unroll
      for (int nt = 0; nt < 4; ++nt)
        acc[nt] = __builtin_amdgcn_wmma_f32_16x16x32_f16(
            false, a, false, b[nt], (short)0, acc[nt], false, false);
    }
    {
      float bvh[8];
#pragma unroll
      for (int v = 0; v < 8; ++v) bvh[v] = sh_bias[3 * DIM + mrow + v];
#pragma unroll
      for (int nt = 0; nt < 4; ++nt) {
        int n_g  = nt * 16 + nlo;
        int base = n_g * DIM + mrow;
#pragma unroll
        for (int v = 0; v < 8; ++v) {
          float hh   = tanhf(acc[nt][v] + bvh[v]);
          float z    = (float)sh_z[base + v];
          float hold = (float)sh_h[base + v];
          float hnew = (1.0f - z) * hold + z * hh;
          sh_h[base + v] = (_Float16)hnew;                  // b128
          sh_hT[(mrow + v) * NODES + n_g] = (_Float16)hnew; // scatter
        }
      }
    }
    __syncthreads();
  }

  float* ho = h_out + (size_t)batch * NODES * DIM;
#pragma unroll 4
  for (int i = tid; i < NODES * DIM; i += 256) ho[i] = (float)sh_h[i];
}

extern "C" void kernel_launch(void* const* d_in, const int* in_sizes, int n_in,
                              void* d_out, int out_size, void* d_ws,
                              size_t ws_size, hipStream_t stream) {
  const float* h_in = (const float*)d_in[0];
  const float* mask = (const float*)d_in[1];
  const float* Wp_w = (const float*)d_in[2];
  const float* Wp_b = (const float*)d_in[3];
  const float* Wz_w = (const float*)d_in[4];
  const float* Wz_b = (const float*)d_in[5];
  const float* Uz_w = (const float*)d_in[6];
  const float* Uz_b = (const float*)d_in[7];
  const float* Wr_w = (const float*)d_in[8];
  const float* Wr_b = (const float*)d_in[9];
  const float* Ur_w = (const float*)d_in[10];
  const float* Ur_b = (const float*)d_in[11];
  const float* Wh_w = (const float*)d_in[12];
  const float* Wh_b = (const float*)d_in[13];
  const float* Uh_w = (const float*)d_in[14];
  const float* Uh_b = (const float*)d_in[15];

  // weights -> f16 in workspace: [Wp, Wz, Uz, Wr, Ur, Wh, Uh]
  _Float16* wf = (_Float16*)d_ws;
  const float* wsrc[7] = {Wp_w, Wz_w, Uz_w, Wr_w, Ur_w, Wh_w, Uh_w};
  for (int i = 0; i < 7; ++i)
    convert_f32_to_f16<<<(DIM * DIM + 255) / 256, 256, 0, stream>>>(
        wsrc[i], wf + i * DIM * DIM, DIM * DIM);

  const int B = in_sizes[1] / (NODES * NODES);  // 4096
  ggsnn_fused<<<B, 256, 0, stream>>>(h_in, mask, wf, Wp_b, Wz_b, Uz_b, Wr_b,
                                     Ur_b, Wh_b, Uh_b, (float*)d_out);
}
